// GPTModel_71717363909255
// MI455X (gfx1250) — matrix operand order repaired and checked by reading, not compiled, not run on hardware
//
#include <hip/hip_runtime.h>
#include <hip/hip_bf16.h>
#include <stdint.h>

// Problem constants (match reference)
#define BB   4
#define TT   1024
#define CCH  768
#define HH   12
#define LL   6
#define VV   50257
#define HSZ  64          // head size
#define FFD  3072        // 4*C
#define MMR  (BB*TT)     // 4096 rows

#define AS1 __attribute__((address_space(1)))
#define AS3 __attribute__((address_space(3)))

typedef __attribute__((ext_vector_type(16))) __bf16  v16bf;
typedef __attribute__((ext_vector_type(8)))  float    v8f;
typedef __attribute__((ext_vector_type(8)))  unsigned v8u;
typedef __attribute__((__vector_size__(16))) int      v4i_t;  // matches builtin param

// CDNA5 async global->LDS copy (ASYNCcnt-tracked), probed so the build stays
// green on toolchains that do not expose the builtin.
#if defined(__has_builtin)
#if __has_builtin(__builtin_amdgcn_global_load_async_to_lds_b128)
#define USE_ASYNC_LDS 1
#endif
#endif

static __device__ __forceinline__ void async_wait0() {
#if defined(USE_ASYNC_LDS)
#if __has_builtin(__builtin_amdgcn_s_wait_asynccnt)
  __builtin_amdgcn_s_wait_asynccnt(0);
#else
  asm volatile("s_wait_asynccnt 0" ::: "memory");
#endif
#endif
}

static __device__ __forceinline__ v16bf as_v16bf(v8u u) {
  return __builtin_bit_cast(v16bf, u);
}
static __device__ __forceinline__ unsigned short bf16_bits(float f) {
  return __builtin_bit_cast(unsigned short, (__bf16)f);
}

// ---------------------------------------------------------------------------
// Embedding: x[b,t,:] = tok_emb[idx[b,t],:] + pos_emb[t,:]
// ---------------------------------------------------------------------------
__global__ void embed_kernel(const int* __restrict__ idx,
                             const float* __restrict__ tok,
                             const float* __restrict__ pos,
                             float* __restrict__ x) {
  int i = blockIdx.x * blockDim.x + threadIdx.x;
  if (i >= MMR * CCH) return;
  int row = i / CCH, c = i % CCH;
  int t = row % TT;
  int token = idx[row];
  x[i] = tok[(size_t)token * CCH + c] + pos[(size_t)t * CCH + c];
}

// ---------------------------------------------------------------------------
// LayerNorm over last dim (C=768), fp32 in -> bf16 out
// ---------------------------------------------------------------------------
__global__ __launch_bounds__(256)
void ln_bf16_kernel(const float* __restrict__ x, const float* __restrict__ g,
                    const float* __restrict__ b, __bf16* __restrict__ out) {
  __shared__ float red[256];
  int row = blockIdx.x, tid = threadIdx.x;
  const float* xr = x + (size_t)row * CCH;
  float s = 0.f, s2 = 0.f;
  for (int c = tid; c < CCH; c += 256) { float v = xr[c]; s += v; s2 += v * v; }
  red[tid] = s; __syncthreads();
  for (int o = 128; o > 0; o >>= 1) { if (tid < o) red[tid] += red[tid + o]; __syncthreads(); }
  float mean = red[0] * (1.0f / CCH);
  __syncthreads();
  red[tid] = s2; __syncthreads();
  for (int o = 128; o > 0; o >>= 1) { if (tid < o) red[tid] += red[tid + o]; __syncthreads(); }
  float var = red[0] * (1.0f / CCH) - mean * mean;
  float rs = rsqrtf(var + 1e-5f);
  for (int c = tid; c < CCH; c += 256)
    out[(size_t)row * CCH + c] = (__bf16)((xr[c] - mean) * rs * g[c] + b[c]);
}

// ---------------------------------------------------------------------------
// Transpose + fp32->bf16: W (K x N, row-major) -> Wt (N x K, row-major)
// blockIdx.z selects layer (stride K*N).
// ---------------------------------------------------------------------------
__global__ void transpose_cvt_kernel(const float* __restrict__ W,
                                     __bf16* __restrict__ Wt, int K, int N) {
  __shared__ float tile[32][33];
  const float* Wl = W + (size_t)blockIdx.z * K * N;
  __bf16* Wtl = Wt + (size_t)blockIdx.z * K * N;
  int n0 = blockIdx.x * 32, k0 = blockIdx.y * 32;
  int tx = threadIdx.x, ty = threadIdx.y;
  #pragma unroll
  for (int i = 0; i < 32; i += 8) {
    int k = k0 + ty + i, n = n0 + tx;
    tile[ty + i][tx] = (k < K && n < N) ? Wl[(size_t)k * N + n] : 0.f;
  }
  __syncthreads();
  #pragma unroll
  for (int i = 0; i < 32; i += 8) {
    int n = n0 + ty + i, k = k0 + tx;
    if (n < N && k < K) Wtl[(size_t)n * K + k] = (__bf16)tile[tx][ty + i];
  }
}

// ---------------------------------------------------------------------------
// WMMA GEMM: out(MxN) = A(MxK,bf16) * Bt(NxK,bf16)^T [+ bias] [relu] [+ res]
// Block tile 128x128, 256 threads (8 waves), K-step 32.
// Wave (wr,wc): wr=wave%4 -> rows 32*wr..+31, wc=wave/4 -> cols 64*wc..+63.
// M multiple of 128, K multiple of 32 (true for all uses); N guarded.
// Async path: double-buffered GLOBAL_LOAD_ASYNC_TO_LDS_B128 pipeline.
// Sync fallback: register-pipelined single-buffer staging.
// ---------------------------------------------------------------------------
template <bool RELU, bool RES, bool OUTBF>
__global__ __launch_bounds__(256)
void gemm_wmma_kernel(const __bf16* __restrict__ A, const __bf16* __restrict__ Bt,
                      const float* __restrict__ bias, const float* __restrict__ res,
                      void* __restrict__ outp, int M, int N, int K) {
#if defined(USE_ASYNC_LDS)
  __shared__ uint4 As4[2][128 * 5];   // 128 rows x 32 bf16, padded to 40 bf16
  __shared__ uint4 Bs4[2][128 * 5];
#else
  __shared__ uint4 As4[1][128 * 5];
  __shared__ uint4 Bs4[1][128 * 5];
#endif

  int tid  = threadIdx.x;
  int lane = tid & 31, wv = tid >> 5;
  int wr = wv & 3, wc = wv >> 2;
  int lm = lane & 15, lh = lane >> 4;
  int mblk = blockIdx.y * 128, nblk = blockIdx.x * 128;
  int srow = tid >> 2, scg = tid & 3;       // staging slot: rows srow, srow+64

  const uint4* Ag = (const uint4*)A;
  const uint4* Bg = (const uint4*)Bt;

  v8f acc[2][4];
  #pragma unroll
  for (int i = 0; i < 2; ++i)
    #pragma unroll
    for (int j = 0; j < 4; ++j)
      #pragma unroll
      for (int e = 0; e < 8; ++e) acc[i][j][e] = 0.f;

  int nk = K >> 5;

#if defined(USE_ASYNC_LDS)
  auto stage_async = [&](int buf, int k0) {
    #pragma unroll
    for (int i = 0; i < 2; ++i) {
      int rr = srow + 64 * i;
      __builtin_amdgcn_global_load_async_to_lds_b128(
          (v4i_t*)(Ag + ((((size_t)(mblk + rr) * K + k0) >> 3) + scg)),
          (AS3 v4i_t*)&As4[buf][rr * 5 + scg], 0, 0);
      int n = nblk + rr;
      if (n < N) {
        __builtin_amdgcn_global_load_async_to_lds_b128(
            (v4i_t*)(Bg + ((((size_t)n * K + k0) >> 3) + scg)),
            (AS3 v4i_t*)&Bs4[buf][rr * 5 + scg], 0, 0);
      } else {
        uint4 z = {0u, 0u, 0u, 0u};
        Bs4[buf][rr * 5 + scg] = z;
      }
    }
  };
  stage_async(0, 0);
#else
  uint4 ra[2], rb[2];
  auto load_regs = [&](int k0) {
    #pragma unroll
    for (int i = 0; i < 2; ++i) {
      int rr = srow + 64 * i;
      ra[i] = Ag[(((size_t)(mblk + rr) * K + k0) >> 3) + scg];
      int n = nblk + rr;
      uint4 z = {0u, 0u, 0u, 0u};
      rb[i] = (n < N) ? Bg[(((size_t)n * K + k0) >> 3) + scg] : z;
    }
  };
  load_regs(0);
#endif

  for (int kt = 0; kt < nk; ++kt) {
#if defined(USE_ASYNC_LDS)
    int cur = kt & 1;
    async_wait0();
    __syncthreads();                         // buf[cur] filled, prior reads done
    if (kt + 1 < nk) stage_async(cur ^ 1, (kt + 1) << 5);
#else
    const int cur = 0;
    #pragma unroll
    for (int i = 0; i < 2; ++i) {
      int rr = srow + 64 * i;
      As4[0][rr * 5 + scg] = ra[i];
      Bs4[0][rr * 5 + scg] = rb[i];
    }
    __syncthreads();
    if (kt + 1 < nk) load_regs((kt + 1) << 5);
#endif
    // L2 prefetch hint two tiles ahead (global_prefetch_b8)
    if (kt + 2 < nk) {
      int k2 = (kt + 2) << 5;
      __builtin_prefetch((const char*)(Ag + ((((size_t)(mblk + srow) * K + k2) >> 3) + scg)), 0, 1);
      int n = nblk + srow;
      if (n < N)
        __builtin_prefetch((const char*)(Bg + ((((size_t)n * K + k2) >> 3) + scg)), 0, 1);
    }

    const unsigned* As = (const unsigned*)As4[cur];
    const unsigned* Bs = (const unsigned*)Bs4[cur];

    // A fragment: lane m = lm, VGPR r holds K pair lh*16 + 2r
    v8u au[2];
    #pragma unroll
    for (int i = 0; i < 2; ++i) {
      int mrow = wr * 32 + i * 16 + lm;
      #pragma unroll
      for (int r = 0; r < 8; ++r)
        au[i][r] = As[mrow * 20 + (lh << 3) + r];
    }
    // B fragment: lane n = lm (+half lh), VGPR r holds K pair lh*16 + 2r
    v8u bu[4];
    #pragma unroll
    for (int j = 0; j < 4; ++j) {
      int nrow = wc * 64 + j * 16 + lm;
      #pragma unroll
      for (int r = 0; r < 8; ++r)
        bu[j][r] = Bs[nrow * 20 + (lh << 3) + r];
    }
    #pragma unroll
    for (int i = 0; i < 2; ++i)
      #pragma unroll
      for (int j = 0; j < 4; ++j)
        acc[i][j] = __builtin_amdgcn_wmma_f32_16x16x32_bf16(
            false, as_v16bf(au[i]), false, as_v16bf(bu[j]),
            (short)0, acc[i][j], false, false);
    __syncthreads();
  }

  // Epilogue. C/D layout: VGPR r -> row r + 8*lh, lane -> col lm.
  #pragma unroll
  for (int j = 0; j < 4; ++j) {
    int cc = nblk + wc * 64 + j * 16 + lm;
    if (cc >= N) continue;
    float bv = bias ? bias[cc] : 0.f;
    #pragma unroll
    for (int i = 0; i < 2; ++i) {
      int rbase = mblk + wr * 32 + i * 16 + lh * 8;
      #pragma unroll
      for (int r = 0; r < 8; ++r) {
        size_t off = (size_t)(rbase + r) * N + cc;
        float v = acc[i][j][r] + bv;
        if (RELU) v = fmaxf(v, 0.f);
        if (RES) v += res[off];
        if (OUTBF) ((__bf16*)outp)[off] = (__bf16)v;
        else       ((float*)outp)[off]  = v;
      }
    }
  }
}

// ---------------------------------------------------------------------------
// Fused causal flash attention, bf16 in/out, fp32 online softmax.
// grid = (T/64, B*H); block = 128 (4 waves); each wave owns a 16-query tile.
// q,k,v are (B*T, C) bf16; head h occupies columns [h*64, h*64+64).
// ---------------------------------------------------------------------------
__global__ __launch_bounds__(128)
void attn_kernel(const __bf16* __restrict__ qB, const __bf16* __restrict__ kB,
                 const __bf16* __restrict__ vB, __bf16* __restrict__ att) {
  __shared__ unsigned Ks[32 * 36];          // 32 keys x 64 hs (dwords, pad 72 bf16)
  __shared__ __bf16 Vts[64 * 40];           // V^T: 64 hs x 32 keys (pad 40)
  __shared__ unsigned short Ps[4 * 16 * 40];// per-wave P: 16 q x 32 keys (pad 40)

  int tid = threadIdx.x, lane = tid & 31, wv = tid >> 5;
  int lm = lane & 15, lh = lane >> 4;
  int qchunk = blockIdx.x;                  // 0..15
  int bh = blockIdx.y;                      // 0..47
  int batch = bh / HH, head = bh % HH;
  int rowb = batch * TT;
  int hcol = head * HSZ;
  int q0 = qchunk * 64 + wv * 16;           // wave's first query (t index)

  const unsigned* qg = (const unsigned*)qB;
  const unsigned* kg = (const unsigned*)kB;
  const unsigned* VtsU = (const unsigned*)Vts;

  // Q fragments (16x64 split into two 16x32 A-fragments), loaded once.
  v8u qa[2];
  #pragma unroll
  for (int f = 0; f < 2; ++f)
    #pragma unroll
    for (int r = 0; r < 8; ++r) {
      int kb = (lh << 4) + (r << 1);
      qa[f][r] = qg[((size_t)(rowb + q0 + lm) * CCH + hcol + f * 32 + kb) >> 1];
    }

  float mrow[8], lrow[8];
  v8f o[4];
  #pragma unroll
  for (int r = 0; r < 8; ++r) { mrow[r] = -3.0e38f; lrow[r] = 0.f; }
  #pragma unroll
  for (int n = 0; n < 4; ++n)
    #pragma unroll
    for (int e = 0; e < 8; ++e) o[n][e] = 0.f;

  const float scale = 0.125f;               // 1/sqrt(64)
  int jmax = 2 * (qchunk + 1);              // 32-key tiles covering causal range

  for (int j = 0; j < jmax; ++j) {
    int jb = j * 32;
    // Stage K tile (32x64) row-major (dword copies).
    #pragma unroll
    for (int i = 0; i < 8; ++i) {
      int linear = tid + 128 * i;
      int row = linear >> 5, dc = linear & 31;
      Ks[row * 36 + dc] = kg[(((size_t)(rowb + jb + row) * CCH + hcol) >> 1) + dc];
    }
    // Stage V tile transposed: Vts[d][kk] = V[jb+kk][hcol+d]
    #pragma unroll
    for (int i = 0; i < 16; ++i) {
      int linear = tid + 128 * i;
      int d = linear & 63, kk = linear >> 6;
      Vts[d * 40 + kk] = vB[(size_t)(rowb + jb + kk) * CCH + hcol + d];
    }
    __syncthreads();

    if (jb <= q0 + 15) {                    // wave-uniform
      // S = Q K^T (two 16x16 tiles over keys), K-step 32 twice over hs.
      v8f s[2];
      #pragma unroll
      for (int nt = 0; nt < 2; ++nt) {
        #pragma unroll
        for (int e = 0; e < 8; ++e) s[nt][e] = 0.f;
        #pragma unroll
        for (int ks = 0; ks < 2; ++ks) {
          v8u bu;
          #pragma unroll
          for (int r = 0; r < 8; ++r)
            bu[r] = Ks[(nt * 16 + lm) * 36 + ks * 16 + (lh << 3) + r];
          s[nt] = __builtin_amdgcn_wmma_f32_16x16x32_bf16(
              false, as_v16bf(qa[ks]), false, as_v16bf(bu),
              (short)0, s[nt], false, false);
        }
      }
      // Scale + causal mask + row max (rows r+8*lh, cols lm within half).
      float mx[8];
      #pragma unroll
      for (int r = 0; r < 8; ++r) mx[r] = -3.0e38f;
      #pragma unroll
      for (int nt = 0; nt < 2; ++nt) {
        int key = jb + nt * 16 + lm;
        #pragma unroll
        for (int r = 0; r < 8; ++r) {
          int qi = q0 + r + 8 * lh;
          float val = s[nt][r] * scale;
          if (key > qi) val = -1.0e30f;
          s[nt][r] = val;
          mx[r] = fmaxf(mx[r], val);
        }
      }
      #pragma unroll
      for (int m = 1; m <= 8; m <<= 1)
        #pragma unroll
        for (int r = 0; r < 8; ++r)
          mx[r] = fmaxf(mx[r], __shfl_xor(mx[r], m, 32));

      float corr[8], ps[8];
      #pragma unroll
      for (int r = 0; r < 8; ++r) {
        float mn = fmaxf(mrow[r], mx[r]);
        corr[r] = __expf(mrow[r] - mn);
        mrow[r] = mn;
      }
      #pragma unroll
      for (int nt = 0; nt < 2; ++nt)
        #pragma unroll
        for (int r = 0; r < 8; ++r)
          s[nt][r] = __expf(s[nt][r] - mrow[r]);
      #pragma unroll
      for (int r = 0; r < 8; ++r) ps[r] = s[0][r] + s[1][r];
      #pragma unroll
      for (int m = 1; m <= 8; m <<= 1)
        #pragma unroll
        for (int r = 0; r < 8; ++r)
          ps[r] += __shfl_xor(ps[r], m, 32);
      #pragma unroll
      for (int r = 0; r < 8; ++r) lrow[r] = lrow[r] * corr[r] + ps[r];
      #pragma unroll
      for (int n = 0; n < 4; ++n)
        #pragma unroll
        for (int r = 0; r < 8; ++r) o[n][r] *= corr[r];

      // C-layout -> A-layout for P via per-wave LDS scratch.
      int wbase = wv * 640;
      #pragma unroll
      for (int nt = 0; nt < 2; ++nt)
        #pragma unroll
        for (int r = 0; r < 8; ++r)
          Ps[wbase + (r + 8 * lh) * 40 + nt * 16 + lm] = bf16_bits(s[nt][r]);
      __threadfence_block();                 // same-wave LDS write->read ordering
      v8u pa;
      #pragma unroll
      for (int r = 0; r < 8; ++r) {
        int col = (lh << 4) + (r << 1);
        unsigned lo = Ps[wbase + lm * 40 + col];
        unsigned hi = Ps[wbase + lm * 40 + col + 1];
        pa[r] = lo | (hi << 16);
      }
      // O += P * V (4 hs tiles of 16)
      #pragma unroll
      for (int n = 0; n < 4; ++n) {
        v8u bu;
        #pragma unroll
        for (int r = 0; r < 8; ++r)
          bu[r] = VtsU[(n * 16 + lm) * 20 + (lh << 3) + r];
        o[n] = __builtin_amdgcn_wmma_f32_16x16x32_bf16(
            false, as_v16bf(pa), false, as_v16bf(bu),
            (short)0, o[n], false, false);
      }
    }
    __syncthreads();
  }

  // Normalize and write bf16 output.
  #pragma unroll
  for (int n = 0; n < 4; ++n)
    #pragma unroll
    for (int r = 0; r < 8; ++r) {
      int t = q0 + r + 8 * lh;
      att[(size_t)(rowb + t) * CCH + hcol + n * 16 + lm] =
          (__bf16)(o[n][r] / lrow[r]);
    }
}

// ---------------------------------------------------------------------------
// Loss: mean over rows of -(log_softmax(logits)[target])
// ---------------------------------------------------------------------------
__global__ void zero_loss_kernel(float* p) { if (threadIdx.x == 0) *p = 0.f; }

__global__ __launch_bounds__(256)
void loss_kernel(const float* __restrict__ logits, const int* __restrict__ targets,
                 float* __restrict__ loss) {
  __shared__ float red[256];
  int row = blockIdx.x, tid = threadIdx.x;
  const float* lr = logits + (size_t)row * VV;
  float mx = -3.0e38f;
  for (int c = tid; c < VV; c += 256) mx = fmaxf(mx, lr[c]);
  red[tid] = mx; __syncthreads();
  for (int o = 128; o > 0; o >>= 1) { if (tid < o) red[tid] = fmaxf(red[tid], red[tid + o]); __syncthreads(); }
  mx = red[0]; __syncthreads();
  float s = 0.f;
  for (int c = tid; c < VV; c += 256) s += __expf(lr[c] - mx);
  red[tid] = s; __syncthreads();
  for (int o = 128; o > 0; o >>= 1) { if (tid < o) red[tid] += red[tid + o]; __syncthreads(); }
  if (tid == 0) {
    float lse = __logf(red[0]) + mx;
    float nll = lse - lr[targets[row]];
    atomicAdd(loss, nll * (1.0f / (float)MMR));
  }
}

// ---------------------------------------------------------------------------
// Host orchestration
// ---------------------------------------------------------------------------
extern "C" void kernel_launch(void* const* d_in, const int* in_sizes, int n_in,
                              void* d_out, int out_size, void* d_ws, size_t ws_size,
                              hipStream_t stream) {
  (void)in_sizes; (void)n_in; (void)out_size; (void)ws_size;
  const int*   idx     = (const int*)d_in[0];
  const int*   targets = (const int*)d_in[1];
  const float* tok_emb = (const float*)d_in[2];
  const float* pos_emb = (const float*)d_in[3];
  const float* Wq = (const float*)d_in[4];
  const float* Wk = (const float*)d_in[5];
  const float* Wv = (const float*)d_in[6];
  const float* Wp = (const float*)d_in[7];
  const float* bp = (const float*)d_in[8];
  const float* ln1_g = (const float*)d_in[9];
  const float* ln1_b = (const float*)d_in[10];
  const float* ln2_g = (const float*)d_in[11];
  const float* ln2_b = (const float*)d_in[12];
  const float* W1 = (const float*)d_in[13];
  const float* b1 = (const float*)d_in[14];
  const float* W2 = (const float*)d_in[15];
  const float* b2 = (const float*)d_in[16];
  const float* lnf_g = (const float*)d_in[17];
  const float* lnf_b = (const float*)d_in[18];
  const float* Wh = (const float*)d_in[19];
  const float* bh = (const float*)d_in[20];

  // Workspace carving (~232 MB total)
  size_t off = 0;
  auto alloc = [&](size_t bytes) -> void* {
    off = (off + 255) & ~(size_t)255;
    void* p = (char*)d_ws + off;
    off += bytes;
    return p;
  };
  float*  x    = (float*)alloc((size_t)MMR * CCH * 4);
  __bf16* hbuf = (__bf16*)alloc((size_t)MMR * CCH * 2);
  __bf16* qb   = (__bf16*)alloc((size_t)MMR * CCH * 2);
  __bf16* kb   = (__bf16*)alloc((size_t)MMR * CCH * 2);
  __bf16* vb   = (__bf16*)alloc((size_t)MMR * CCH * 2);
  __bf16* attb = (__bf16*)alloc((size_t)MMR * CCH * 2);
  __bf16* ffb  = (__bf16*)alloc((size_t)MMR * FFD * 2);
  __bf16* WqT  = (__bf16*)alloc((size_t)LL * CCH * CCH * 2);
  __bf16* WkT  = (__bf16*)alloc((size_t)LL * CCH * CCH * 2);
  __bf16* WvT  = (__bf16*)alloc((size_t)LL * CCH * CCH * 2);
  __bf16* WpT  = (__bf16*)alloc((size_t)LL * CCH * CCH * 2);
  __bf16* W1T  = (__bf16*)alloc((size_t)LL * CCH * FFD * 2);
  __bf16* W2T  = (__bf16*)alloc((size_t)LL * CCH * FFD * 2);
  __bf16* WhT  = (__bf16*)alloc((size_t)CCH * VV * 2);

  float* logits = (float*)d_out;
  float* lossp  = logits + (size_t)MMR * VV;

  auto cdiv = [](int a, int b) { return (a + b - 1) / b; };
  dim3 tb(32, 8);

  // Weight conversion + transpose (bf16, N x K layout)
  transpose_cvt_kernel<<<dim3(cdiv(CCH, 32), cdiv(CCH, 32), LL), tb, 0, stream>>>(Wq, WqT, CCH, CCH);
  transpose_cvt_kernel<<<dim3(cdiv(CCH, 32), cdiv(CCH, 32), LL), tb, 0, stream>>>(Wk, WkT, CCH, CCH);
  transpose_cvt_kernel<<<dim3(cdiv(CCH, 32), cdiv(CCH, 32), LL), tb, 0, stream>>>(Wv, WvT, CCH, CCH);
  transpose_cvt_kernel<<<dim3(cdiv(CCH, 32), cdiv(CCH, 32), LL), tb, 0, stream>>>(Wp, WpT, CCH, CCH);
  transpose_cvt_kernel<<<dim3(cdiv(FFD, 32), cdiv(CCH, 32), LL), tb, 0, stream>>>(W1, W1T, CCH, FFD);
  transpose_cvt_kernel<<<dim3(cdiv(CCH, 32), cdiv(FFD, 32), LL), tb, 0, stream>>>(W2, W2T, FFD, CCH);
  transpose_cvt_kernel<<<dim3(cdiv(VV, 32),  cdiv(CCH, 32), 1),  tb, 0, stream>>>(Wh, WhT, CCH, VV);

  // Embedding
  embed_kernel<<<cdiv(MMR * CCH, 256), 256, 0, stream>>>(idx, tok_emb, pos_emb, x);

  for (int l = 0; l < LL; ++l) {
    const __bf16* wqT = WqT + (size_t)l * CCH * CCH;
    const __bf16* wkT = WkT + (size_t)l * CCH * CCH;
    const __bf16* wvT = WvT + (size_t)l * CCH * CCH;
    const __bf16* wpT = WpT + (size_t)l * CCH * CCH;
    const __bf16* w1T = W1T + (size_t)l * CCH * FFD;
    const __bf16* w2T = W2T + (size_t)l * CCH * FFD;

    ln_bf16_kernel<<<MMR, 256, 0, stream>>>(x, ln1_g + l * CCH, ln1_b + l * CCH, hbuf);

    dim3 gcc(cdiv(CCH, 128), MMR / 128);
    gemm_wmma_kernel<false, false, true><<<gcc, 256, 0, stream>>>(hbuf, wqT, nullptr, nullptr, qb, MMR, CCH, CCH);
    gemm_wmma_kernel<false, false, true><<<gcc, 256, 0, stream>>>(hbuf, wkT, nullptr, nullptr, kb, MMR, CCH, CCH);
    gemm_wmma_kernel<false, false, true><<<gcc, 256, 0, stream>>>(hbuf, wvT, nullptr, nullptr, vb, MMR, CCH, CCH);

    attn_kernel<<<dim3(TT / 64, BB * HH), 128, 0, stream>>>(qb, kb, vb, attb);

    gemm_wmma_kernel<false, true, false><<<gcc, 256, 0, stream>>>(attb, wpT, bp + l * CCH, x, x, MMR, CCH, CCH);

    ln_bf16_kernel<<<MMR, 256, 0, stream>>>(x, ln2_g + l * CCH, ln2_b + l * CCH, hbuf);

    dim3 gff(cdiv(FFD, 128), MMR / 128);
    gemm_wmma_kernel<true, false, true><<<gff, 256, 0, stream>>>(hbuf, w1T, b1 + l * FFD, nullptr, ffb, MMR, FFD, CCH);
    gemm_wmma_kernel<false, true, false><<<gcc, 256, 0, stream>>>(ffb, w2T, b2 + l * CCH, x, x, MMR, CCH, FFD);
  }

  // Final LN + LM head into d_out logits
  ln_bf16_kernel<<<MMR, 256, 0, stream>>>(x, lnf_g, lnf_b, hbuf);
  dim3 gv(cdiv(VV, 128), MMR / 128);
  gemm_wmma_kernel<false, false, false><<<gv, 256, 0, stream>>>(hbuf, WhT, bh, nullptr, logits, MMR, VV, CCH);

  // Loss
  zero_loss_kernel<<<1, 32, 0, stream>>>(lossp);
  loss_kernel<<<MMR, 256, 0, stream>>>(logits, targets, lossp);
}
